// WindowAttention_15564961481372
// MI455X (gfx1250) — matrix operand order repaired and checked by reading, not compiled, not run on hardware
//
#include <hip/hip_runtime.h>
#include <hip/hip_bf16.h>

// ---------------------------------------------------------------------------
// Swin window attention, MI455X (gfx1250, wave32, WMMA bf16 16x16x32,
// async global->LDS staging).
//   DIM=512, HEADS=16, HD=32, N=64 tokens/window, B=2048 windows, NW=64.
// ---------------------------------------------------------------------------

typedef __bf16 bf16;
typedef __attribute__((ext_vector_type(16))) __bf16 bf16x16;
typedef __attribute__((ext_vector_type(8)))  float  f32x8;
typedef __attribute__((ext_vector_type(4)))  int    v4i;

#define WMMA_BF16(A, B, C)                                                     \
  __builtin_amdgcn_wmma_f32_16x16x32_bf16(false, (A), false, (B), (short)0,    \
                                          (C), false, false)

static constexpr int kDim   = 512;
static constexpr int kHeads = 16;
static constexpr int kHd    = 32;
static constexpr int kN     = 64;        // tokens per window
static constexpr int kB     = 2048;      // windows
static constexpr int kRows  = kB * kN;   // 131072
static constexpr int kQKV   = 3 * kDim;  // 1536
static constexpr float kScale = 0.17677669529663687f; // 32^-0.5

// ---------------------------------------------------------------------------
// Async global->LDS copy (16B per lane). GLOBAL_LOAD_ASYNC_TO_LDS_B128 is
// tracked by ASYNCcnt and bypasses VGPRs entirely; fall back to a VGPR
// round-trip if the builtin is unavailable on this toolchain.
// Builtin signature (from hipcc diagnostic): param0 is
// '__device__ (AS1) int4*', so cast through ext-vector int4 pointers.
// ---------------------------------------------------------------------------
#if __has_builtin(__builtin_amdgcn_global_load_async_to_lds_b128)
#define HAVE_ASYNC_LDS 1
__device__ __forceinline__ void async_copy16(const void* g, void* l) {
  __builtin_amdgcn_global_load_async_to_lds_b128(
      (__attribute__((address_space(1))) v4i*)(g),
      (__attribute__((address_space(3))) v4i*)(l), 0, 0);
}
#else
#define HAVE_ASYNC_LDS 0
__device__ __forceinline__ void async_copy16(const void* g, void* l) {
  *reinterpret_cast<uint4*>(l) = *reinterpret_cast<const uint4*>(g);
}
#endif

__device__ __forceinline__ void wait_async() {
#if HAVE_ASYNC_LDS
#if __has_builtin(__builtin_amdgcn_s_wait_asynccnt)
  __builtin_amdgcn_s_wait_asynccnt(0);
#else
  asm volatile("s_wait_asynccnt 0" ::: "memory");
#endif
#endif
}

// ---------------------------------------------------------------------------
// Fragment loaders. Matrices are addressed as row-major [rows][ld] bf16.
// A-frag (16x32, MxK): lane holds row = lane&15; elems 0..7  -> K = h*8+0..7,
//                      elems 8..15 -> K = 16+h*8+0..7   (h = lane>>4).
// B-frag (32x16, KxN) read from an [N][K] array: lane holds N = lane&15,
//                      elems 0..7 -> K = h*8+0..7, elems 8..15 -> K = 16+h*8+0..7.
// All offsets are 16B aligned when ld*2 is a multiple of 16 (ld=40,72,512,1536).
// ---------------------------------------------------------------------------
__device__ __forceinline__ bf16x16 load_a_frag(const bf16* __restrict__ row0,
                                               int ld, int lane) {
  int r = lane & 15, h = lane >> 4;
  const bf16* p = row0 + (size_t)r * ld + h * 8;
  union { bf16x16 v; uint4 q[2]; } u;
  u.q[0] = *reinterpret_cast<const uint4*>(p);
  u.q[1] = *reinterpret_cast<const uint4*>(p + 16);
  return u.v;
}

__device__ __forceinline__ bf16x16 load_b_frag(const bf16* __restrict__ n0,
                                               int ld, int lane) {
  int n = lane & 15, h = lane >> 4;
  const bf16* p = n0 + (size_t)n * ld + h * 8;
  union { bf16x16 v; uint4 q[2]; } u;
  u.q[0] = *reinterpret_cast<const uint4*>(p);
  u.q[1] = *reinterpret_cast<const uint4*>(p + 16);
  return u.v;
}

// ---------------------------------------------------------------------------
// fp32 -> bf16 convert (weights)
// ---------------------------------------------------------------------------
__global__ void cvt_f32_bf16(const float* __restrict__ s, bf16* __restrict__ d,
                             int n) {
  int i = blockIdx.x * blockDim.x + threadIdx.x;
  if (i < n) d[i] = (bf16)s[i];
}

// ---------------------------------------------------------------------------
// Tiled WMMA GEMM:  C[M][Nc] = A[M][K] * W[Nc][K]^T + bias[Nc]
// Block tile 128x128, K-step 32, 256 threads = 8 waves (4x2 wave grid,
// each wave owns a 32x64 sub-tile = 2x4 WMMA accumulators).
// Weight tiles (and bf16 A tiles) are staged with async global->LDS DMA,
// double-buffered and overlapped with the WMMA compute of the live stage.
// A_F32: A is fp32 and converted to bf16 on a VGPR staging path.
// C_F32: epilogue writes fp32 (else bf16).
// ---------------------------------------------------------------------------
template <bool A_F32, bool C_F32>
__global__ __launch_bounds__(256) void gemm_wmma(
    const void* __restrict__ Aptr, const bf16* __restrict__ W,
    const float* __restrict__ bias, void* __restrict__ Cptr, int M, int Nc,
    int K) {
  constexpr int LDK = 40;  // 32 + 8 pad: keeps 16B align, staggers banks
  __shared__ bf16 sA[2][128][LDK];
  __shared__ bf16 sB[2][128][LDK];

  const int tid  = threadIdx.x;
  const int lane = tid & 31;
  const int wid  = tid >> 5;
  const int wm   = wid & 3;   // 0..3 -> 32-row strip
  const int wn   = wid >> 2;  // 0..1 -> 64-col strip
  const int m0   = blockIdx.x * 128;
  const int n0   = blockIdx.y * 128;

  const int lrow = tid >> 1;          // 0..127
  const int lcol = (tid & 1) * 16;    // 0 or 16

  f32x8 acc[2][4];
#pragma unroll
  for (int i = 0; i < 2; ++i)
#pragma unroll
    for (int j = 0; j < 4; ++j) acc[i][j] = (f32x8)0.0f;

  const int nk = K / 32;

  auto load_tiles = [&](int kc, int stg) {
    const int k0 = kc * 32;
    // --- A tile ---
    if (A_F32) {
      const float* A = (const float*)Aptr;
      const float4* pa = reinterpret_cast<const float4*>(
          A + (size_t)(m0 + lrow) * K + k0 + lcol);
      float4 f0 = pa[0], f1 = pa[1], f2 = pa[2], f3 = pa[3];
      union { bf16 e[16]; uint4 q[2]; } t;
      t.e[0] = (bf16)f0.x;  t.e[1] = (bf16)f0.y;
      t.e[2] = (bf16)f0.z;  t.e[3] = (bf16)f0.w;
      t.e[4] = (bf16)f1.x;  t.e[5] = (bf16)f1.y;
      t.e[6] = (bf16)f1.z;  t.e[7] = (bf16)f1.w;
      t.e[8] = (bf16)f2.x;  t.e[9] = (bf16)f2.y;
      t.e[10] = (bf16)f2.z; t.e[11] = (bf16)f2.w;
      t.e[12] = (bf16)f3.x; t.e[13] = (bf16)f3.y;
      t.e[14] = (bf16)f3.z; t.e[15] = (bf16)f3.w;
      uint4* d = reinterpret_cast<uint4*>(&sA[stg][lrow][lcol]);
      d[0] = t.q[0];
      d[1] = t.q[1];
    } else {
      const bf16* A = (const bf16*)Aptr;
      const bf16* ga = A + (size_t)(m0 + lrow) * K + k0 + lcol;
      bf16* la = &sA[stg][lrow][lcol];
      async_copy16(ga, la);
      async_copy16(ga + 8, la + 8);
    }
    // --- B (weight) tile, W is [Nc][K] ---
    const bf16* gb = W + (size_t)(n0 + lrow) * K + k0 + lcol;
    bf16* lb = &sB[stg][lrow][lcol];
    async_copy16(gb, lb);
    async_copy16(gb + 8, lb + 8);
  };

  load_tiles(0, 0);
  wait_async();
  __syncthreads();

  for (int kc = 0; kc < nk; ++kc) {
    const int cur = kc & 1;
    if (kc + 1 < nk) load_tiles(kc + 1, cur ^ 1);  // DMA overlaps WMMA below

    bf16x16 af[2], bfr[4];
#pragma unroll
    for (int i = 0; i < 2; ++i)
      af[i] = load_a_frag(&sA[cur][wm * 32 + i * 16][0], LDK, lane);
#pragma unroll
    for (int j = 0; j < 4; ++j)
      bfr[j] = load_b_frag(&sB[cur][wn * 64 + j * 16][0], LDK, lane);
#pragma unroll
    for (int i = 0; i < 2; ++i)
#pragma unroll
      for (int j = 0; j < 4; ++j)
        acc[i][j] = WMMA_BF16(af[i], bfr[j], acc[i][j]);

    wait_async();
    __syncthreads();
  }

  // epilogue: D layout -> row = s + (lane>>4)*8, col = lane&15 per 16x16 tile
  const int rh = (lane >> 4) * 8;
  const int cl = lane & 15;
#pragma unroll
  for (int i = 0; i < 2; ++i) {
#pragma unroll
    for (int j = 0; j < 4; ++j) {
      const int c = n0 + wn * 64 + j * 16 + cl;
      const float bv = bias[c];
      const int mbase = m0 + wm * 32 + i * 16 + rh;
#pragma unroll
      for (int s = 0; s < 8; ++s) {
        const float v = acc[i][j][s] + bv;
        const size_t off = (size_t)(mbase + s) * Nc + c;
        if (C_F32)
          ((float*)Cptr)[off] = v;
        else
          ((bf16*)Cptr)[off] = (bf16)v;
      }
    }
  }
}

// ---------------------------------------------------------------------------
// Attention: 128 threads = 4 waves; wave handles one (window b, head h).
// grid = (2048, 4); h = blockIdx.y*4 + wid.
//   scores = (q k^T)*scale + bias[rel_idx] + mask[b%64]   (16 WMMAs)
//   softmax rows via width-16 shfl reductions (the two 16-lane halves of a
//   wave32 hold different M rows of the D tile, so width-16 is the row domain)
//   out = P v                                              (16 WMMAs)
// P staged in LDS (A-frag shape), v staged transposed in LDS (B-frag shape).
// ---------------------------------------------------------------------------
__global__ __launch_bounds__(128) void attn_wmma(
    const bf16* __restrict__ qkv, const float* __restrict__ bias_table,
    const int* __restrict__ rel_idx, const float* __restrict__ mask,
    bf16* __restrict__ outp) {
  constexpr int LDA = 72;  // 64 + 8 pad (144B pitch, 16B aligned)
  __shared__ bf16 sAttn[4][kN][LDA];
  __shared__ bf16 sVT[4][kHd][LDA];

  const int lane = threadIdx.x & 31;
  const int wid  = threadIdx.x >> 5;
  const int b    = blockIdx.x;
  const int h    = blockIdx.y * 4 + wid;
  const int w    = b & 63;  // mask window index (b % NW)

  const size_t rowB = (size_t)b * kN * kQKV;  // window base in qkv
  const int hq = h * kHd;          // q col offset
  const int hk = kDim + h * kHd;   // k col offset
  const int hv = 2 * kDim + h * kHd;

  // ---- stage v transposed: sVT[d][j] = v[j][d].
  // Lane owns columns j = 2*lane, 2*lane+1 -> one packed b32 store per row.
  {
    const int j0 = lane * 2;
    union { bf16 e[32]; uint4 q[4]; } v0, v1;
    const uint4* p0 =
        reinterpret_cast<const uint4*>(qkv + rowB + (size_t)j0 * kQKV + hv);
    const uint4* p1 = reinterpret_cast<const uint4*>(
        qkv + rowB + (size_t)(j0 + 1) * kQKV + hv);
#pragma unroll
    for (int t = 0; t < 4; ++t) {
      v0.q[t] = p0[t];
      v1.q[t] = p1[t];
    }
#pragma unroll
    for (int d = 0; d < kHd; ++d) {
      union { bf16 e[2]; unsigned u; } pk;
      pk.e[0] = v0.e[d];
      pk.e[1] = v1.e[d];
      *reinterpret_cast<unsigned*>(&sVT[wid][d][j0]) = pk.u;
    }
  }

  // ---- k fragments (B operand of q*k^T): 4 tiles of 16 tokens ----
  bf16x16 kf[4];
#pragma unroll
  for (int nj = 0; nj < 4; ++nj)
    kf[nj] = load_b_frag(qkv + rowB + (size_t)(nj * 16) * kQKV + hk, kQKV, lane);

  const int rh = (lane >> 4) * 8;
  const int cl = lane & 15;

  // ---- scores + softmax, one 16-row block at a time ----
#pragma unroll
  for (int mi = 0; mi < 4; ++mi) {
    bf16x16 qf =
        load_a_frag(qkv + rowB + (size_t)(mi * 16) * kQKV + hq, kQKV, lane);
    f32x8 acc[4];
#pragma unroll
    for (int nj = 0; nj < 4; ++nj) acc[nj] = (f32x8)0.0f;
#pragma unroll
    for (int nj = 0; nj < 4; ++nj) acc[nj] = WMMA_BF16(qf, kf[nj], acc[nj]);

#pragma unroll
    for (int s = 0; s < 8; ++s) {
      const int i = mi * 16 + rh + s;  // global query row
      float v[4];
#pragma unroll
      for (int nj = 0; nj < 4; ++nj) {
        const int j = nj * 16 + cl;
        const int idx = rel_idx[i * kN + j];
        const float bv = bias_table[idx * kHeads + h];
        const float mv = mask[((size_t)w * kN + i) * kN + j];
        v[nj] = acc[nj][s] * kScale + bv + mv;
      }
      float rmax = fmaxf(fmaxf(v[0], v[1]), fmaxf(v[2], v[3]));
#pragma unroll
      for (int m = 1; m < 16; m <<= 1)
        rmax = fmaxf(rmax, __shfl_xor(rmax, m, 16));
      float e[4], rsum = 0.f;
#pragma unroll
      for (int nj = 0; nj < 4; ++nj) {
        e[nj] = __expf(v[nj] - rmax);
        rsum += e[nj];
      }
#pragma unroll
      for (int m = 1; m < 16; m <<= 1) rsum += __shfl_xor(rsum, m, 16);
      const float inv = 1.f / rsum;
#pragma unroll
      for (int nj = 0; nj < 4; ++nj)
        sAttn[wid][i][nj * 16 + cl] = (bf16)(e[nj] * inv);
    }
  }
  // per-wave LDS RAW: DS ops from one wave are in-order (DScnt), and waves use
  // disjoint LDS regions -> no workgroup barrier needed.

  // ---- out = P @ v : B frags from sVT [d][j] ----
  bf16x16 bv[2][2];
#pragma unroll
  for (int nd = 0; nd < 2; ++nd)
#pragma unroll
    for (int kc = 0; kc < 2; ++kc)
      bv[nd][kc] = load_b_frag(&sVT[wid][nd * 16][kc * 32], LDA, lane);

#pragma unroll
  for (int mi = 0; mi < 4; ++mi) {
    f32x8 acc[2];
    acc[0] = (f32x8)0.0f;
    acc[1] = (f32x8)0.0f;
#pragma unroll
    for (int kc = 0; kc < 2; ++kc) {
      bf16x16 af = load_a_frag(&sAttn[wid][mi * 16][kc * 32], LDA, lane);
#pragma unroll
      for (int nd = 0; nd < 2; ++nd) acc[nd] = WMMA_BF16(af, bv[nd][kc], acc[nd]);
    }
#pragma unroll
    for (int nd = 0; nd < 2; ++nd) {
#pragma unroll
      for (int s = 0; s < 8; ++s) {
        const int n = mi * 16 + rh + s;
        const int d = nd * 16 + cl;
        outp[((size_t)b * kN + n) * kDim + h * kHd + d] = (bf16)acc[nd][s];
      }
    }
  }
}

// ---------------------------------------------------------------------------
// Launch. Inputs: x, mask, qkv_w, qkv_b, proj_w, proj_b, bias_table, rel_idx.
// Workspace layout (bf16): qkv_w | proj_w | qkv[131072][1536] | att[131072][512]
// ---------------------------------------------------------------------------
extern "C" void kernel_launch(void* const* d_in, const int* in_sizes, int n_in,
                              void* d_out, int out_size, void* d_ws,
                              size_t ws_size, hipStream_t stream) {
  const float* x          = (const float*)d_in[0];
  const float* mask       = (const float*)d_in[1];
  const float* qkv_w      = (const float*)d_in[2];
  const float* qkv_b      = (const float*)d_in[3];
  const float* proj_w     = (const float*)d_in[4];
  const float* proj_b     = (const float*)d_in[5];
  const float* bias_table = (const float*)d_in[6];
  const int*   rel_idx    = (const int*)d_in[7];
  float* out = (float*)d_out;

  char* ws = (char*)d_ws;
  bf16* qkvw_bf = (bf16*)(ws);
  bf16* projw_bf = (bf16*)(ws + (size_t)kQKV * kDim * 2);
  bf16* qkv_bf = (bf16*)(ws + (size_t)kQKV * kDim * 2 + (size_t)kDim * kDim * 2);
  bf16* att_bf = (bf16*)((char*)qkv_bf + (size_t)kRows * kQKV * 2);

  // 1) weight conversion
  {
    int n1 = kQKV * kDim;  // 786432
    cvt_f32_bf16<<<(n1 + 255) / 256, 256, 0, stream>>>(qkv_w, qkvw_bf, n1);
    int n2 = kDim * kDim;  // 262144
    cvt_f32_bf16<<<(n2 + 255) / 256, 256, 0, stream>>>(proj_w, projw_bf, n2);
  }

  // 2) QKV projection: [131072,512] x [1536,512]^T -> bf16 [131072,1536]
  gemm_wmma<true, false><<<dim3(kRows / 128, kQKV / 128), 256, 0, stream>>>(
      (const void*)x, qkvw_bf, qkv_b, (void*)qkv_bf, kRows, kQKV, kDim);

  // 3) attention per (window, head)
  attn_wmma<<<dim3(kB, 4), 128, 0, stream>>>(qkv_bf, bias_table, rel_idx, mask,
                                             att_bf);

  // 4) output projection: [131072,512] x [512,512]^T -> fp32 d_out
  gemm_wmma<false, true><<<dim3(kRows / 128, kDim / 128), 256, 0, stream>>>(
      (const void*)att_bf, projw_bf, proj_b, (void*)out, kRows, kDim, kDim);
}
